// BasisConv_18305150616437
// MI455X (gfx1250) — matrix operand order repaired and checked by reading, not compiled, not run on hardware
//
#include <hip/hip_runtime.h>

typedef __attribute__((ext_vector_type(16))) _Float16 v16h;
typedef __attribute__((ext_vector_type(8)))  float    v8f;

#define F_IN  32
#define F_OUT 32
#define NBAS  4
#define KB    16               // NBAS*NBAS basis terms
#define TILE_E 16              // edges (M rows) per wave
#define WAVES_PER_BLOCK 4
#define BLOCK_THREADS (WAVES_PER_BLOCK * 32)

// LDS B-tiles in WMMA layout: [kb][ntile][lane][16 halves] -> 16*2*32*16 halves = 32 KB
#define WLDS_HALFS (KB * 2 * 32 * 16)

__global__ void zero_f32_kernel(float* __restrict__ p, int n) {
    int i = blockIdx.x * blockDim.x + threadIdx.x;
    if (i < n) p[i] = 0.0f;
}

__global__ __launch_bounds__(BLOCK_THREADS)
void basisconv_edge_wmma(const float* __restrict__ x_j,
                         const int*   __restrict__ edge_index,  // [2,E]: row0=dst, row1=src
                         const float* __restrict__ edge_attr,   // [E,2]
                         const float* __restrict__ weight,      // [4,4,32,32] fp32
                         float*       __restrict__ out,         // [N,32]
                         int E)
{
    __shared__ __align__(32) _Float16 wlds[WLDS_HALFS];

    // ---- Stage W (512x32) into LDS, pre-swizzled into WMMA B-layout (f16) ----
    // For k-step kb, N-tile nt: lane ln holds column o = nt*16 + (ln&15),
    // halves j=0..15 holding rows K = (ln&16) + j  (K local to this kb-step == feature i).
    for (int idx = threadIdx.x; idx < WLDS_HALFS; idx += BLOCK_THREADS) {
        int kb = idx >> 10;           // /1024
        int r  = idx & 1023;
        int nt = r >> 9;              // /512
        int r2 = r & 511;
        int ln = r2 >> 4;             // lane 0..31
        int j  = r2 & 15;             // half index within lane
        int i  = (ln & 16) + j;       // feature row (K within step)
        int o  = nt * 16 + (ln & 15); // output column
        wlds[idx] = (_Float16)weight[(kb * F_IN + i) * F_OUT + o];
    }
    __syncthreads();

    int wave = threadIdx.x >> 5;
    int lane = threadIdx.x & 31;
    long tile = (long)blockIdx.x * WAVES_PER_BLOCK + wave;
    long ebase = tile * TILE_E;
    if (ebase >= E) return;          // uniform per wave

    // Lanes L and L+16 both own edge (ebase + L&15) -> row M = L&15 of the A tile.
    int em    = (int)ebase + (lane & 15);
    bool valid = em < E;
    int emc   = valid ? em : (E - 1);

    int   dst = edge_index[emc];
    int   src = edge_index[E + emc];
    float ex  = edge_attr[2 * emc];
    float ey  = edge_attr[2 * emc + 1];

    // Hat basis on [-1,1], 4 centers: c_t = -1 + t*(2/3), f = max(0, 1 - |x-c|*1.5)
    float bx[NBAS], by[NBAS];
#pragma unroll
    for (int t = 0; t < NBAS; t++) {
        float c = -1.0f + (float)t * (2.0f / 3.0f);
        bx[t] = fmaxf(0.0f, 1.0f - fabsf(ex - c) * 1.5f);
        by[t] = fmaxf(0.0f, 1.0f - fabsf(ey - c) * 1.5f);
    }
    if (!valid) {                    // zero contribution; keeps EXEC full for WMMA
#pragma unroll
        for (int t = 0; t < NBAS; t++) bx[t] = 0.0f;
    }

    // Pre-convert all 16 basis products to f16 once (outside the WMMA loop).
    _Float16 bh[KB];
#pragma unroll
    for (int kb = 0; kb < KB; kb++)
        bh[kb] = (_Float16)(bx[kb >> 2] * by[kb & 3]);

    // ---- Gather feature row into a single v16h in WMMA A-layout ----
    // low half-wave (lane<16): K = 0..15; high half-wave: K = 16..31
    int ibase = (lane & 16) ? 16 : 0;
    const float* fp = x_j + (long)src * F_IN;
    float4 f0 = *(const float4*)(fp + ibase);          // K = ibase..ibase+3
    float4 f1 = *(const float4*)(fp + ibase + 4);      // K = ibase+4..ibase+7
    float4 f2 = *(const float4*)(fp + ibase + 8);      // K = ibase+8..
    float4 f3 = *(const float4*)(fp + ibase + 12);

    v16h fav;
    fav[0]  = (_Float16)f0.x; fav[1]  = (_Float16)f0.y;
    fav[2]  = (_Float16)f0.z; fav[3]  = (_Float16)f0.w;
    fav[4]  = (_Float16)f1.x; fav[5]  = (_Float16)f1.y;
    fav[6]  = (_Float16)f1.z; fav[7]  = (_Float16)f1.w;
    fav[8]  = (_Float16)f2.x; fav[9]  = (_Float16)f2.y;
    fav[10] = (_Float16)f2.z; fav[11] = (_Float16)f2.w;
    fav[12] = (_Float16)f3.x; fav[13] = (_Float16)f3.y;
    fav[14] = (_Float16)f3.z; fav[15] = (_Float16)f3.w;

    v8f acc0 = {};  // columns o = 0..15
    v8f acc1 = {};  // columns o = 16..31

    // ---- K loop: 16 basis steps x (K=32) ; A_step = feat row scaled by b[e,kb] ----
#pragma unroll
    for (int kb = 0; kb < KB; kb++) {
        v16h a = fav * bh[kb];      // scalar splat -> 8x v_pk_mul_f16

        const v16h* bp = (const v16h*)(wlds + kb * 1024);
        v16h b0 = bp[lane];         // N-tile 0
        v16h b1 = bp[32 + lane];    // N-tile 1

        acc0 = __builtin_amdgcn_wmma_f32_16x16x32_f16(
            false, a, false, b0, (short)0, acc0, false, false);
        acc1 = __builtin_amdgcn_wmma_f32_16x16x32_f16(
            false, a, false, b1, (short)0, acc1, false, false);
    }

    // ---- Scatter-add: C layout lane L, VGPR v -> M = v + 8*(L>=16), N = L&15 ----
    int orow = lane & 15;
    int moff = (lane >> 4) << 3;
#pragma unroll
    for (int v = 0; v < 8; v++) {
        int m = moff + v;
        int d = __shfl(dst, m, 32);          // lane m owns edge m's dst
        atomicAdd(out + (long)d * F_OUT + orow,      acc0[v]);
        atomicAdd(out + (long)d * F_OUT + 16 + orow, acc1[v]);
    }
}

extern "C" void kernel_launch(void* const* d_in, const int* in_sizes, int n_in,
                              void* d_out, int out_size, void* d_ws, size_t ws_size,
                              hipStream_t stream) {
    (void)d_ws; (void)ws_size; (void)n_in;
    // inputs: x_i, x_j, edge_index, edge_attr, weight
    const float* x_j       = (const float*)d_in[1];
    const int*   edge_idx  = (const int*)  d_in[2];
    const float* edge_attr = (const float*)d_in[3];
    const float* weight    = (const float*)d_in[4];
    float*       out       = (float*)d_out;

    int E = in_sizes[3] / 2;

    // zero the output accumulator (harness poisons d_out)
    int zthreads = 256;
    int zblocks  = (out_size + zthreads - 1) / zthreads;
    zero_f32_kernel<<<zblocks, zthreads, 0, stream>>>(out, out_size);

    long tiles  = ((long)E + TILE_E - 1) / TILE_E;
    long blocks = (tiles + WAVES_PER_BLOCK - 1) / WAVES_PER_BLOCK;
    basisconv_edge_wmma<<<(int)blocks, BLOCK_THREADS, 0, stream>>>(
        x_j, edge_idx, edge_attr, weight, out, E);
}
